// VectorQuantizerEMA_46883863003322
// MI455X (gfx1250) — matrix operand order repaired and checked by reading, not compiled, not run on hardware
//
#include <hip/hip_runtime.h>

typedef __attribute__((ext_vector_type(16))) __bf16 v16bf;
typedef __attribute__((ext_vector_type(8)))  float  v8f;

constexpr int D = 64;            // embedding dim
constexpr int K = 512;           // codebook size
constexpr int THREADS = 256;     // 8 wave32 per block
constexpr int WAVES = THREADS / 32;
constexpr int ROWS_PER_BLOCK = WAVES * 16;  // 128 rows per block
constexpr int SROW = 72;         // LDS row stride in ushorts (144 B, bank-conflict-free)

static __device__ __forceinline__ unsigned short f2bf(float f) {
    unsigned u = __builtin_bit_cast(unsigned int, f);
    unsigned r = (u + 0x7FFFu + ((u >> 16) & 1u)) >> 16;   // round-to-nearest-even
    return (unsigned short)r;
}

// ---------------- Kernel 1: codebook -> bf16 + ||c||^2 ----------------
__global__ void vq_precompute(const float* __restrict__ cw,
                              unsigned short* __restrict__ cw_bf,
                              float* __restrict__ csq) {
    int n = blockIdx.x * blockDim.x + threadIdx.x;
    if (n < K) {
        float s = 0.f;
        #pragma unroll 4
        for (int d = 0; d < D; ++d) {
            float v = cw[n * D + d];
            s += v * v;
            cw_bf[n * D + d] = f2bf(v);
        }
        csq[n] = s;
    }
}

// ---------------- Kernel 2: distances via WMMA + argmin + gather ----------------
__global__ void __launch_bounds__(THREADS)
vq_main(const float* __restrict__ inp, const float* __restrict__ cw,
        const unsigned short* __restrict__ cw_bf, const float* __restrict__ csq,
        float* __restrict__ out_q, float* __restrict__ out_idx,
        float* __restrict__ partials) {
    __shared__ __align__(16) unsigned short s_cw[K * SROW];  // 72 KB padded bf16 codebook
    __shared__ float s_csq[K];                               // 2 KB
    __shared__ int   s_sel[WAVES * 16];                      // chosen code per row

    const int tid = threadIdx.x;

    // Stage codebook into LDS with padded row stride (128-bit chunks)
    {
        const uint4* src = (const uint4*)cw_bf;              // 8 chunks of 16B per row
        for (int i = tid; i < K * 8; i += THREADS) {
            const int row = i >> 3;
            const int c   = i & 7;
            ((uint4*)(s_cw + row * SROW))[c] = src[i];
        }
        for (int i = tid; i < K; i += THREADS) s_csq[i] = csq[i];
    }
    __syncthreads();

    const int wave = tid >> 5;
    const int lane = tid & 31;
    const int hsel = lane >> 4;          // which 16-lane half
    const int l16  = lane & 15;
    const long mbase = (long)blockIdx.x * ROWS_PER_BLOCK + wave * 16;

    // ---- Load A operands (16x32 bf16 per k-chunk), per ISA A layout ----
    // lanes 0-15:  row M=l16, K = kc*32 + {0..7} (elts 0-7) and {16..23} (elts 8-15)
    // lanes 16-31: row M=l16, K = kc*32 + {8..15} and {24..31}
    v16bf a0, a1;
    {
        const float* rp = inp + (mbase + l16) * D;
        #pragma unroll
        for (int kc = 0; kc < 2; ++kc) {
            const int k0 = kc * 32 + hsel * 8;
            float4 f0 = *(const float4*)(rp + k0);
            float4 f1 = *(const float4*)(rp + k0 + 4);
            float4 f2 = *(const float4*)(rp + k0 + 16);
            float4 f3 = *(const float4*)(rp + k0 + 20);
            union { unsigned short s[16]; v16bf v; } u;
            u.s[0]  = f2bf(f0.x); u.s[1]  = f2bf(f0.y); u.s[2]  = f2bf(f0.z); u.s[3]  = f2bf(f0.w);
            u.s[4]  = f2bf(f1.x); u.s[5]  = f2bf(f1.y); u.s[6]  = f2bf(f1.z); u.s[7]  = f2bf(f1.w);
            u.s[8]  = f2bf(f2.x); u.s[9]  = f2bf(f2.y); u.s[10] = f2bf(f2.z); u.s[11] = f2bf(f2.w);
            u.s[12] = f2bf(f3.x); u.s[13] = f2bf(f3.y); u.s[14] = f2bf(f3.z); u.s[15] = f2bf(f3.w);
            if (kc == 0) a0 = u.v; else a1 = u.v;
        }
    }

    // ---- Sweep all 32 column tiles of the codebook ----
    float best[8];
    int   bidx[8];
    #pragma unroll
    for (int j = 0; j < 8; ++j) { best[j] = 3.4e38f; bidx[j] = 0; }

    #pragma unroll 2
    for (int t = 0; t < K / 16; ++t) {
        const int n = t * 16 + l16;       // this lane's column (matches C layout)
        // B (32x16 bf16): lanes 0-15 hold K 0..7,16..23 of column n, lanes 16-31 hold K 8..15,24..31
        const unsigned short* bp = s_cw + n * SROW + hsel * 8;
        union { uint4 q[2]; v16bf v; } b0u, b1u;
        b0u.q[0] = *(const uint4*)(bp);
        b0u.q[1] = *(const uint4*)(bp + 16);
        b1u.q[0] = *(const uint4*)(bp + 32);
        b1u.q[1] = *(const uint4*)(bp + 48);
        // Independent accumulators: no WMMA->WMMA RAW hazard, both pipeline
        v8f c0 = {}, c1 = {};
        c0 = __builtin_amdgcn_wmma_f32_16x16x32_bf16(false, a0, false, b0u.v, (short)0, c0, false, false);
        c1 = __builtin_amdgcn_wmma_f32_16x16x32_bf16(false, a1, false, b1u.v, (short)0, c1, false, false);
        const float bias = s_csq[n];
        #pragma unroll
        for (int j = 0; j < 8; ++j) {
            // relative distance: ||c||^2 - 2 x.c  (x^2 is constant per row)
            float s    = c0[j] + c1[j];
            float dist = __builtin_fmaf(-2.0f, s, bias);
            if (dist < best[j]) { best[j] = dist; bidx[j] = n; }   // n strictly increasing per lane
        }
    }

    // ---- Cross-lane argmin within each 16-lane group (rows j / j+8) ----
    #pragma unroll
    for (int j = 0; j < 8; ++j) {
        #pragma unroll
        for (int off = 1; off < 16; off <<= 1) {
            float ob = __shfl_xor(best[j], off, 16);
            int   oi = __shfl_xor(bidx[j], off, 16);
            // branchless tie-break: smaller value, then smaller index
            int take = (int)(ob < best[j]) | ((int)(ob == best[j]) & (int)(oi < bidx[j]));
            best[j] = take ? ob : best[j];
            bidx[j] = take ? oi : bidx[j];
        }
    }
    if (l16 == 0) {
        #pragma unroll
        for (int j = 0; j < 8; ++j) s_sel[wave * 16 + hsel * 8 + j] = bidx[j];
    }
    // same-wave LDS ops are in-order; all lanes of this wave execute the reads below

    // ---- Gather codewords, write outputs, exact fp32 commitment error ----
    float err = 0.f;
    const int d2 = lane * 2;   // 32 lanes x 2 floats = 64
    for (int r = 0; r < 16; ++r) {
        const long m = mbase + r;
        const int  n = s_sel[wave * 16 + r];
        if (lane == 0) out_idx[m] = (float)n;
        const float2 xv = *(const float2*)(inp + m * D + d2);
        const float2 cv = *(const float2*)(cw + (long)n * D + d2);
        *(float2*)(out_q + m * D + d2) = cv;
        float dx = cv.x - xv.x, dy = cv.y - xv.y;
        err += dx * dx + dy * dy;
    }
    #pragma unroll
    for (int off = 1; off < 32; off <<= 1) err += __shfl_xor(err, off, 32);
    if (lane == 0) partials[blockIdx.x * WAVES + wave] = err;
}

// ---------------- Kernel 3: deterministic fixed-order loss reduction ----------------
__global__ void vq_finalize(const float* __restrict__ partials, float* __restrict__ loss,
                            int nPart, float scale) {
    __shared__ float red[256];
    float s = 0.f;
    for (int i = threadIdx.x; i < nPart; i += 256) s += partials[i];
    red[threadIdx.x] = s;
    __syncthreads();
    for (int w = 128; w > 0; w >>= 1) {
        if (threadIdx.x < w) red[threadIdx.x] += red[threadIdx.x + w];
        __syncthreads();
    }
    if (threadIdx.x == 0) loss[0] = red[0] * scale;
}

extern "C" void kernel_launch(void* const* d_in, const int* in_sizes, int n_in,
                              void* d_out, int out_size, void* d_ws, size_t ws_size,
                              hipStream_t stream) {
    const float* inp = (const float*)d_in[0];    // (64,4096,64) fp32
    const float* cw  = (const float*)d_in[1];    // (512,64) fp32
    const long Ntot = (long)in_sizes[0] / D;     // 262144 rows

    float* out      = (float*)d_out;
    float* out_q    = out;                       // Ntot*D floats
    float* out_idx  = out + Ntot * (long)D;      // Ntot floats (indices as float)
    float* out_loss = out_idx + Ntot;            // 1 float

    unsigned short* cw_bf = (unsigned short*)d_ws;                         // 64 KB
    float* csq      = (float*)((char*)d_ws + (size_t)K * D * 2);           // 2 KB
    float* partials = (float*)((char*)d_ws + (size_t)K * D * 2 + K * 4);   // 64 KB

    const int nBlocks = (int)(Ntot / ROWS_PER_BLOCK);                      // 2048

    vq_precompute<<<(K + 255) / 256, 256, 0, stream>>>(cw, cw_bf, csq);
    vq_main<<<nBlocks, THREADS, 0, stream>>>(inp, cw, cw_bf, csq, out_q, out_idx, partials);
    vq_finalize<<<1, 256, 0, stream>>>(partials, out_loss, nBlocks * WAVES,
                                       0.25f / ((float)Ntot * (float)D));
}
